// Attention_19662360281297
// MI455X (gfx1250) — matrix operand order repaired and checked by reading, not compiled, not run on hardware
//
#include <hip/hip_runtime.h>

// ---------------- problem constants (match reference) ----------------
#define BB   8
#define TT   16
#define DD   1024
#define NN   16
#define KHn  8
#define HH   128
#define SS   8192
#define GGrp (NN / KHn)
#define EPSI 1e-6f
#define LNEG (-3.3895313892515355e+38f)   // finfo(bf16).min

typedef __attribute__((ext_vector_type(16))) __bf16        v16bf;
typedef __attribute__((ext_vector_type(8)))  float         v8f;
typedef __attribute__((ext_vector_type(4)))  float         v4f;
typedef __attribute__((ext_vector_type(4)))  unsigned int  v4u;

typedef __attribute__((ext_vector_type(4))) unsigned int u32x4;
typedef __attribute__((ext_vector_type(8))) int          i32x8;
typedef __attribute__((ext_vector_type(4))) int          i32x4;

union BFU { v16bf v; unsigned short u[16]; v4u q[2]; };

// f32 -> bf16 via the hardware conversion path (RNE), not integer emulation
__device__ __forceinline__ unsigned short f2bf(float f) {
  return __builtin_bit_cast(unsigned short, (__bf16)f);
}

__device__ __forceinline__ v8f vzero8() {
  v8f z;
#pragma unroll
  for (int i = 0; i < 8; ++i) z[i] = 0.f;
  return z;
}

__device__ __forceinline__ float redmax16(float v) {
  v = fmaxf(v, __shfl_xor(v, 1));
  v = fmaxf(v, __shfl_xor(v, 2));
  v = fmaxf(v, __shfl_xor(v, 4));
  v = fmaxf(v, __shfl_xor(v, 8));
  return v;
}
__device__ __forceinline__ float redadd16(float v) {
  v += __shfl_xor(v, 1); v += __shfl_xor(v, 2);
  v += __shfl_xor(v, 4); v += __shfl_xor(v, 8);
  return v;
}
__device__ __forceinline__ float redadd32(float v) {
  v = redadd16(v); v += __shfl_xor(v, 16); return v;
}

// WMMA A operand (16x32 bf16): lane l holds row M=l&15;
// lanes 0-15 carry K=0..15, lanes 16-31 K=16..31.
__device__ __forceinline__ void loadA(BFU& a, const float* chunk, int akb) {
  v4f f0 = *(const v4f*)(chunk + akb);
  v4f f1 = *(const v4f*)(chunk + akb + 4);
  v4f f2 = *(const v4f*)(chunk + akb + 8);
  v4f f3 = *(const v4f*)(chunk + akb + 12);
#pragma unroll
  for (int j = 0; j < 4; ++j) {
    a.u[j]      = f2bf(f0[j]);
    a.u[4 + j]  = f2bf(f1[j]);
    a.u[8 + j]  = f2bf(f2[j]);
    a.u[12 + j] = f2bf(f3[j]);
  }
}

// WMMA B operand (32x16 bf16) when this lane's 16 K-values are contiguous.
__device__ __forceinline__ void load16(BFU& b, const float* p) {
  v4f f0 = *(const v4f*)(p);
  v4f f1 = *(const v4f*)(p + 4);
  v4f f2 = *(const v4f*)(p + 8);
  v4f f3 = *(const v4f*)(p + 12);
#pragma unroll
  for (int j = 0; j < 4; ++j) {
    b.u[j]      = f2bf(f0[j]);
    b.u[4 + j]  = f2bf(f1[j]);
    b.u[8 + j]  = f2bf(f2[j]);
    b.u[12 + j] = f2bf(f3[j]);
  }
}

__device__ __forceinline__ v8f wmma_bf16(const BFU& a, const BFU& b, v8f c) {
  return __builtin_amdgcn_wmma_f32_16x16x32_bf16(false, a.v, false, b.v,
                                                 (short)0, c, false, false);
}

// ---------------- Tensor Data Mover: 2D tile (rows x HH f32) -> LDS ----------------
__device__ __forceinline__ void tdm_load_tile(unsigned lds_byte, const float* gsrc,
                                              unsigned rows, unsigned row_stride_elems) {
  unsigned long long ga = (unsigned long long)(uintptr_t)gsrc;
  u32x4 g0;
  g0[0] = 1u;                                                  // count=1, user descriptor
  g0[1] = lds_byte;                                            // D#.lds_addr
  g0[2] = (unsigned)ga;                                        // global_addr[31:0]
  g0[3] = (unsigned)((ga >> 32) & 0x01ffffffu) | (2u << 30);   // addr[56:32] | type=2
  unsigned td0 = HH, td1 = rows, t0 = HH, t1 = rows;
  i32x8 g1;
  g1[0] = (int)(2u << 16);                                     // data_size = 4B
  g1[1] = (int)((td0 & 0xffffu) << 16);                        // tensor_dim0[15:0] @bit48
  g1[2] = (int)(((td0 >> 16) & 0xffffu) | ((td1 & 0xffffu) << 16));   // td0 hi | td1 lo
  g1[3] = (int)(((td1 >> 16) & 0xffffu) | ((t0 & 0xffffu) << 16));    // td1 hi | tile_dim0
  g1[4] = (int)(t1 & 0xffffu);                                 // tile_dim1 (tile_dim2=0)
  g1[5] = (int)row_stride_elems;                               // tensor_dim0_stride lo32
  g1[6] = 0;
  g1[7] = 0;
  i32x4 gz; gz[0] = 0; gz[1] = 0; gz[2] = 0; gz[3] = 0;        // groups 2/3 unused (2D)
#if defined(__clang_major__) && (__clang_major__ >= 23)
  i32x8 gz8;
#pragma unroll
  for (int i = 0; i < 8; ++i) gz8[i] = 0;
  __builtin_amdgcn_tensor_load_to_lds(g0, g1, gz, gz, gz8, 0);
#else
  __builtin_amdgcn_tensor_load_to_lds(g0, g1, gz, gz, 0);
#endif
}

// 16x16 GEMM tile: per-lane A row pointer, B column pointer (stride = bstride).
__device__ __forceinline__ v8f gemm16(const float* arow, const float* bcol,
                                      int bstride, int kdim) {
  int l   = threadIdx.x;
  int akb = (l < 16) ? 0 : 16;
  int bkb = (l < 16) ? 0 : 16;
  v8f acc = vzero8();
  for (int kc = 0; kc < kdim / 32; ++kc) {
    BFU a, b;
    loadA(a, arow + kc * 32, akb);
#pragma unroll
    for (int e = 0; e < 16; ++e)
      b.u[e] = f2bf(bcol[(kc * 32 + bkb + e) * bstride]);
    acc = wmma_bf16(a, b, acc);
  }
  return acc;
}

// ---------------- kernel 0: per-batch meta (start, first-nonzero) ----------------
__global__ void prep_meta(const int* __restrict__ seg,
                          const int* __restrict__ start_ind,
                          int* __restrict__ meta) {
  int b = threadIdx.x;
  if (b >= BB) return;
  const int* row = seg + b * TT;
  int left = 0;
  while (left < TT && row[left] == 0) ++left;
  int first = (left < TT) ? left : 0;          // argmax of (row != 0)
  int st = start_ind[b];
  if (st < 0) st = left;
  meta[b]      = st;
  meta[BB + b] = first;
}

// ---------------- kernel 1: fused QKV projection (bf16 WMMA) ----------------
__global__ void gemm_qkv(const float* __restrict__ x,
                         const float* __restrict__ Wq,
                         const float* __restrict__ Wk,
                         const float* __restrict__ Wv,
                         float* __restrict__ qbuf,
                         float* __restrict__ kbuf,
                         float* __restrict__ vbuf) {
  int rt = blockIdx.x;
  int c0 = blockIdx.y * 16;
  int l  = threadIdx.x;

  const float* W; int wstride; int cloc; float* dst; int dstride;
  if (c0 < NN * HH)              { W = Wq; wstride = NN * HH;  cloc = c0;                    dst = qbuf; dstride = NN * HH; }
  else if (c0 < NN*HH + KHn*HH)  { W = Wk; wstride = KHn * HH; cloc = c0 - NN*HH;            dst = kbuf; dstride = KHn * HH; }
  else                           { W = Wv; wstride = KHn * HH; cloc = c0 - NN*HH - KHn*HH;   dst = vbuf; dstride = KHn * HH; }

  int col = cloc + (l & 15);
  const float* arow = x + (rt * 16 + (l & 15)) * DD;
  v8f acc = gemm16(arow, W + col, wstride, DD);

  int hi = l >> 4;
#pragma unroll
  for (int r = 0; r < 8; ++r)
    dst[(rt * 16 + hi * 8 + r) * dstride + col] = acc[r];
}

// ---------------- kernel 2: RMSNorm + scale + RoPE, in place ----------------
__global__ void norm_rope(float* __restrict__ qbuf, float* __restrict__ kbuf,
                          const float* __restrict__ q_scale,
                          const float* __restrict__ k_scale,
                          const int* __restrict__ seg,
                          const int* __restrict__ meta,
                          const int* __restrict__ curp) {
  int id = blockIdx.x;
  int hh = id % (NN + KHn);
  int bt = id / (NN + KHn);
  int b = bt / TT, t = bt % TT;
  int l = threadIdx.x;

  float* row; const float* sc;
  if (hh < NN) { row = qbuf + (bt * NN + hh) * HH;         sc = q_scale; }
  else         { row = kbuf + (bt * KHn + (hh - NN)) * HH; sc = k_scale; }

  v4f v = *(const v4f*)(row + l * 4);
  float ss = v[0]*v[0] + v[1]*v[1] + v[2]*v[2] + v[3]*v[3];
  ss = redadd32(ss);
  float rinv = rsqrtf(ss * (1.0f / (float)HH) + EPSI);

  int cur = curp[0];
  int first = meta[BB + b];
  float pos = (seg[b * TT + t] != 0) ? (float)(t - first + cur)
                                     : ((float)(1 << 30) + (float)cur);
  float outv[4];
#pragma unroll
  for (int j = 0; j < 4; ++j) {
    int h = l * 4 + j;
    float y = v[j] * rinv * sc[h];
    float partner = __shfl_xor(y, 16);           // value at h +/- 64
    int i = (h < 64) ? h : (h - 64);
    float invf = __expf(-((float)(2 * i) * (1.0f / (float)HH)) * 13.815510557964274f); // theta=1e6
    float ang = pos * invf, s, c;
    __sincosf(ang, &s, &c);
    outv[j] = (h < 64) ? (y * c - partner * s) : (y * c + partner * s);
  }
  v4f o; o[0]=outv[0]; o[1]=outv[1]; o[2]=outv[2]; o[3]=outv[3];
  *(v4f*)(row + l * 4) = o;
}

// ---------------- kernel 3: flash attention, one wave per (b, q-head) ----------------
// K blocks inside the cache region are streamed by the Tensor Data Mover into a
// double-buffered LDS tile (issue next / wait tensorcnt<=1 / consume current).
__global__ void attention(const float* __restrict__ qbuf,
                          const float* __restrict__ kbuf,
                          const float* __restrict__ vbuf,
                          const float* __restrict__ k_cache,
                          const float* __restrict__ v_cache,
                          const int* __restrict__ seg,
                          const int* __restrict__ meta,
                          const int* __restrict__ curp,
                          float* __restrict__ obuf) {
  __shared__ float Kt[2][32][HH];                // 32 KB TDM double buffer
  __shared__ unsigned short Pl[16][32];          // P tile, bf16 bits

  int b  = blockIdx.x / NN;
  int n  = blockIdx.x % NN;
  int kh = n / GGrp;
  int cur = curp[0];
  int start = meta[b];
  int slimit = cur + TT;

  int l  = threadIdx.x;
  int lo = l & 15, hi = l >> 4;
  int akb = hi ? 16 : 0;                         // A-operand K base
  int kbk = hi ? 16 : 0;                         // B-operand K base

  const float* qrow = qbuf + ((b * TT + lo) * NN + n) * HH;
  BFU qa[4];
#pragma unroll
  for (int hc = 0; hc < 4; ++hc) loadA(qa[hc], qrow + hc * 32, akb);

  int segv[8];
#pragma unroll
  for (int r = 0; r < 8; ++r) segv[r] = seg[b * TT + hi * 8 + r];

  float mrow[8], lrow[8];
  v8f oacc[8];
#pragma unroll
  for (int r = 0; r < 8; ++r) { mrow[r] = -1e38f; lrow[r] = 0.f; }
#pragma unroll
  for (int j = 0; j < 8; ++j) oacc[j] = vzero8();

  const float scl = 0.08838834764831845f;        // H^-0.5

  unsigned ktoff0 = (unsigned)(uintptr_t)&Kt[0][0][0];
  unsigned ktoff1 = (unsigned)(uintptr_t)&Kt[1][0][0];

  int sbeg = start & ~31;
  int par = 0;
  bool pendCur = false;
  if (sbeg + 32 <= cur) {                        // full block inside cache region
    tdm_load_tile(ktoff0, k_cache + ((b * SS + sbeg) * KHn + kh) * HH, 32u, KHn * HH);
    pendCur = true;
  }

  for (int s0 = sbeg; s0 < slimit; s0 += 32) {
    // ---- software pipeline: issue TDM for the next block ----
    bool pendNext = false;
    int nx = s0 + 32;
    if (nx < slimit && nx + 32 <= cur) {
      asm volatile("s_wait_dscnt 0" ::: "memory");   // buffer reads from 2 blocks ago done
      tdm_load_tile(par ? ktoff0 : ktoff1,
                    k_cache + ((b * SS + nx) * KHn + kh) * HH, 32u, KHn * HH);
      pendNext = true;
    }

    // ---- logits: two 16-wide tiles ----
    v8f c[2]; c[0] = vzero8(); c[1] = vzero8();
    if (pendCur) {
      if (pendNext) __builtin_amdgcn_s_wait_tensorcnt(1);
      else          __builtin_amdgcn_s_wait_tensorcnt(0);
#pragma unroll
      for (int u = 0; u < 2; ++u) {
        const float* kr = &Kt[par][u * 16 + lo][0];
#pragma unroll
        for (int hc = 0; hc < 4; ++hc) {
          BFU kb;
          load16(kb, kr + hc * 32 + kbk);
          c[u] = wmma_bf16(qa[hc], kb, c[u]);
        }
      }
    } else {
      // boundary block: mixes cache rows and freshly-projected K from workspace
#pragma unroll
      for (int u = 0; u < 2; ++u) {
        int s = s0 + u * 16 + lo;
        int sc_ = (s < slimit) ? s : (slimit - 1); if (sc_ < 0) sc_ = 0;
        const float* krow = (sc_ < cur)
            ? (k_cache + ((b * SS + sc_) * KHn + kh) * HH)
            : (kbuf    + ((b * TT + (sc_ - cur)) * KHn + kh) * HH);
#pragma unroll
        for (int hc = 0; hc < 4; ++hc) {
          BFU kb;
          load16(kb, krow + hc * 32 + kbk);
          c[u] = wmma_bf16(qa[hc], kb, c[u]);
        }
      }
    }

    // ---- mask + online softmax (rows live across 16-lane groups) ----
    float p0[8], p1[8], alpha[8];
#pragma unroll
    for (int r = 0; r < 8; ++r) {
      int tt = hi * 8 + r;
      float lg[2];
#pragma unroll
      for (int u = 0; u < 2; ++u) {
        int s = s0 + u * 16 + lo;
        int kvseg = (s >= start && s < slimit) ? 1 : 0;
        bool valid = (s >= 0) && (s < slimit) && (s <= cur + tt) && (kvseg == segv[r]);
        lg[u] = valid ? c[u][r] * scl : LNEG;
      }
      float mx   = redmax16(fmaxf(lg[0], lg[1]));
      float mnew = fmaxf(mrow[r], mx);
      alpha[r] = __expf(fmaxf(mrow[r] - mnew, -88.f));
      p0[r] = (lg[0] < -1e37f) ? 0.f : __expf(lg[0] - mnew);
      p1[r] = (lg[1] < -1e37f) ? 0.f : __expf(lg[1] - mnew);
      float sum = redadd16(p0[r] + p1[r]);
      mrow[r] = mnew;
      lrow[r] = lrow[r] * alpha[r] + sum;
    }
#pragma unroll
    for (int j = 0; j < 8; ++j)
#pragma unroll
      for (int r = 0; r < 8; ++r) oacc[j][r] *= alpha[r];

    // ---- stage P through LDS: C layout -> A layout ----
#pragma unroll
    for (int r = 0; r < 8; ++r) {
      int tt = hi * 8 + r;
      Pl[tt][lo]      = f2bf(p0[r]);
      Pl[tt][16 + lo] = f2bf(p1[r]);
    }
    asm volatile("s_wait_dscnt 0" ::: "memory");
    BFU pa;
    pa.q[0] = *(const v4u*)&Pl[lo][akb];
    pa.q[1] = *(const v4u*)&Pl[lo][akb + 8];

    // ---- P (16x32) x V (32x16 per h-chunk) ----
#pragma unroll
    for (int j = 0; j < 8; ++j) {
      BFU vb;
      int hcol = j * 16 + lo;
#pragma unroll
      for (int e = 0; e < 16; ++e) {
        int s = s0 + kbk + e;
        int sc_ = (s < slimit) ? s : (slimit - 1); if (sc_ < 0) sc_ = 0;
        const float* vp = (sc_ < cur)
            ? (v_cache + ((b * SS + sc_) * KHn + kh) * HH + hcol)
            : (vbuf    + ((b * TT + (sc_ - cur)) * KHn + kh) * HH + hcol);
        vb.u[e] = f2bf(*vp);
      }
      oacc[j] = wmma_bf16(pa, vb, oacc[j]);
    }

    par ^= 1;
    pendCur = pendNext;
  }

  // ---- epilogue: normalize and store o ----
  float rcp[8];
#pragma unroll
  for (int r = 0; r < 8; ++r) rcp[r] = (lrow[r] > 0.f) ? (1.f / lrow[r]) : 0.f;
#pragma unroll
  for (int j = 0; j < 8; ++j)
#pragma unroll
    for (int r = 0; r < 8; ++r) {
      int tt = hi * 8 + r;
      obuf[((b * TT + tt) * NN + n) * HH + j * 16 + lo] = oacc[j][r] * rcp[r];
    }
}

// ---------------- kernel 4: output projection (bf16 WMMA) ----------------
__global__ void out_proj(const float* __restrict__ obuf,
                         const float* __restrict__ Wo,
                         float* __restrict__ out) {
  int rt = blockIdx.x;
  int c0 = blockIdx.y * 16;
  int l  = threadIdx.x;
  int col = c0 + (l & 15);
  const float* arow = obuf + (rt * 16 + (l & 15)) * (NN * HH);
  v8f acc = gemm16(arow, Wo + col, DD, NN * HH);
  int hi = l >> 4;
#pragma unroll
  for (int r = 0; r < 8; ++r)
    out[(rt * 16 + hi * 8 + r) * DD + col] = acc[r];
}

// ---------------- launch ----------------
extern "C" void kernel_launch(void* const* d_in, const int* in_sizes, int n_in,
                              void* d_out, int out_size, void* d_ws, size_t ws_size,
                              hipStream_t stream) {
  (void)in_sizes; (void)n_in; (void)out_size; (void)ws_size;
  const float* x        = (const float*)d_in[0];
  const float* Wq       = (const float*)d_in[1];
  const float* Wk       = (const float*)d_in[2];
  const float* Wv       = (const float*)d_in[3];
  const float* Wo       = (const float*)d_in[4];
  const float* q_scale  = (const float*)d_in[5];
  const float* k_scale  = (const float*)d_in[6];
  const float* k_cache  = (const float*)d_in[7];
  const float* v_cache  = (const float*)d_in[8];
  const int*   seg      = (const int*)d_in[9];
  const int*   startp   = (const int*)d_in[10];
  const int*   curp     = (const int*)d_in[11];

  float* ws   = (float*)d_ws;
  float* qbuf = ws;                                     // B*T*N*H
  float* kbuf = qbuf + (size_t)BB * TT * NN * HH;       // B*T*K*H
  float* vbuf = kbuf + (size_t)BB * TT * KHn * HH;      // B*T*K*H
  float* obuf = vbuf + (size_t)BB * TT * KHn * HH;      // B*T*N*H
  int*   meta = (int*)(obuf + (size_t)BB * TT * NN * HH);

  prep_meta<<<1, 32, 0, stream>>>(seg, startp, meta);
  gemm_qkv<<<dim3(BB, (NN * HH + 2 * KHn * HH) / 16), 32, 0, stream>>>(
      x, Wq, Wk, Wv, qbuf, kbuf, vbuf);
  norm_rope<<<BB * TT * (NN + KHn), 32, 0, stream>>>(
      qbuf, kbuf, q_scale, k_scale, seg, meta, curp);
  attention<<<BB * NN, 32, 0, stream>>>(
      qbuf, kbuf, vbuf, k_cache, v_cache, seg, meta, curp, obuf);
  out_proj<<<dim3(BB, DD / 16), 32, 0, stream>>>(obuf, Wo, (float*)d_out);
}
